// PlasticBitLinear_28389733826762
// MI455X (gfx1250) — matrix operand order repaired and checked by reading, not compiled, not run on hardware
//
#include <hip/hip_runtime.h>
#include <math.h>

// ---------------------------------------------------------------------------
// PlasticBitLinear on MI455X (gfx1250):
//   y      = (x @ ternary(W)^T) * scale + bias          (4096x4096 * 4096x16384)
//   traceO = trace*0.95 + (y^T @ x)/tokens              (16384x4096)
// Both GEMMs run on v_wmma_f32_16x16x32_f16 with f32 accumulation, with
// software-pipelined double-buffered LDS (global prefetch into registers,
// one barrier per K-step).
// ---------------------------------------------------------------------------

typedef _Float16 v16h __attribute__((ext_vector_type(16)));
typedef _Float16 v8h  __attribute__((ext_vector_type(8)));
typedef _Float16 v4h  __attribute__((ext_vector_type(4)));
typedef float    v8f  __attribute__((ext_vector_type(8)));

#define TOKENS 4096
#define DIN    4096
#define DOUT   16384
#define TK     32
#define LDA    40   // 32 + 8 halves pad -> 80B rows, keeps every frag read 16B aligned

// ---- WMMA fragment loaders (per ISA 7.12.2 VGPR layouts, wave32) ------------

// A (16x32 f16): lanes 0-15 -> M=lane, halves 0..7 = K 0..7, halves 8..15 = K 16..23
//                lanes 16-31 -> M=lane-16, K offset by +8 in each group.
static __device__ __forceinline__ v16h frag_a(const _Float16* rowp, int grp) {
  const v8h lo = *(const v8h*)(rowp + grp * 8);
  const v8h hi = *(const v8h*)(rowp + grp * 8 + 16);
  v16h a;
#pragma unroll
  for (int j = 0; j < 8; ++j) { a[j] = lo[j]; a[j + 8] = hi[j]; }
  return a;
}

// B (32x16 f16): lane selects column N; same K order as A (halves 0..7 = K 0..7, 8..15 = K 16..23, +8 for lanes 16-31).
// LDS stores B transposed as [N][K] so this is 2 contiguous b128 reads.
static __device__ __forceinline__ v16h frag_b(const _Float16* colp, int grp) {
  const v8h lo = *(const v8h*)(colp + grp * 8);
  const v8h hi = *(const v8h*)(colp + grp * 8 + 16);
  v16h b;
#pragma unroll
  for (int j = 0; j < 8; ++j) { b[j] = lo[j]; b[j + 8] = hi[j]; }
  return b;
}

// 8 WMMAs on one 128x128 block tile step (wave owns a 64x32 sub-tile).
static __device__ __forceinline__ void wmma_step(const _Float16* As, const _Float16* Bs,
                                                 int wm, int wn, int lane_lo, int grp,
                                                 v8f acc[4][2]) {
  v16h bf[2];
#pragma unroll
  for (int nt = 0; nt < 2; ++nt)
    bf[nt] = frag_b(Bs + (wn * 32 + nt * 16 + lane_lo) * LDA, grp);
#pragma unroll
  for (int mt = 0; mt < 4; ++mt) {
    const v16h af = frag_a(As + (wm * 64 + mt * 16 + lane_lo) * LDA, grp);
#pragma unroll
    for (int nt = 0; nt < 2; ++nt)
      acc[mt][nt] = __builtin_amdgcn_wmma_f32_16x16x32_f16(
          false, af, false, bf[nt], (short)0, acc[mt][nt], false, false);
  }
}

// ---- Deterministic reduction for thr = 0.1 * mean(|W|) ----------------------

__global__ __launch_bounds__(256) void absum_partial(const float* __restrict__ w,
                                                     float* __restrict__ partials) {
  __shared__ float sdata[256];
  const int tid = threadIdx.x;
  const size_t base = (size_t)blockIdx.x * 16384;
  float s = 0.f;
#pragma unroll
  for (int i = 0; i < 16; ++i) {
    const float4 v = *(const float4*)(w + base + (size_t)i * 1024 + tid * 4);
    s += fabsf(v.x) + fabsf(v.y) + fabsf(v.z) + fabsf(v.w);
  }
  sdata[tid] = s;
  __syncthreads();
  for (int off = 128; off > 0; off >>= 1) {
    if (tid < off) sdata[tid] += sdata[tid + off];
    __syncthreads();
  }
  if (tid == 0) partials[blockIdx.x] = sdata[0];
}

__global__ __launch_bounds__(256) void absum_final(const float* __restrict__ partials,
                                                   float* __restrict__ thr_out) {
  __shared__ float sdata[256];
  const int tid = threadIdx.x;
  float s = 0.f;
  for (int i = tid; i < 4096; i += 256) s += partials[i];
  sdata[tid] = s;
  __syncthreads();
  for (int off = 128; off > 0; off >>= 1) {
    if (tid < off) sdata[tid] += sdata[tid + off];
    __syncthreads();
  }
  if (tid == 0) thr_out[0] = 0.1f * sdata[0] / (float)((size_t)DOUT * DIN);
}

// ---- GEMM1: y[t,o] = sum_i x[t,i]*ternary(W)[o,i] * scale + bias[o] ---------

static __device__ __forceinline__ void g1_load(const float* __restrict__ x,
                                               const float* __restrict__ w,
                                               int m0, int n0, int kk, int tid,
                                               float4 pa[4], float4 pb[4]) {
#pragma unroll
  for (int p = 0; p < 4; ++p) {
    const int idx = p * 1024 + tid * 4;
    const int r = idx >> 5, c = idx & 31;
    pa[p] = *(const float4*)(x + (size_t)(m0 + r) * DIN + kk + c);
    pb[p] = *(const float4*)(w + (size_t)(n0 + r) * DIN + kk + c);
  }
}

static __device__ __forceinline__ void g1_store(_Float16* As, _Float16* Bs, float thr,
                                                int tid, const float4 pa[4],
                                                const float4 pb[4]) {
#pragma unroll
  for (int p = 0; p < 4; ++p) {
    const int idx = p * 1024 + tid * 4;
    const int r = idx >> 5, c = idx & 31;
    const float4 a = pa[p];
    v4h ha; ha[0] = (_Float16)a.x; ha[1] = (_Float16)a.y; ha[2] = (_Float16)a.z; ha[3] = (_Float16)a.w;
    *(v4h*)(As + r * LDA + c) = ha;
    const float4 b = pb[p];
    v4h hb;
    hb[0] = (_Float16)((fabsf(b.x) > thr) ? (b.x > 0.f ? 1.f : -1.f) : 0.f);
    hb[1] = (_Float16)((fabsf(b.y) > thr) ? (b.y > 0.f ? 1.f : -1.f) : 0.f);
    hb[2] = (_Float16)((fabsf(b.z) > thr) ? (b.z > 0.f ? 1.f : -1.f) : 0.f);
    hb[3] = (_Float16)((fabsf(b.w) > thr) ? (b.w > 0.f ? 1.f : -1.f) : 0.f);
    *(v4h*)(Bs + r * LDA + c) = hb;
  }
}

__global__ __launch_bounds__(256)
void gemm1_ternary(const float* __restrict__ x, const float* __restrict__ w,
                   const float* __restrict__ bias, const float* __restrict__ wscale,
                   const float* __restrict__ thrp, float* __restrict__ y) {
  __shared__ _Float16 As[2][128 * LDA];
  __shared__ _Float16 Bs[2][128 * LDA];
  const int tid = threadIdx.x;
  const int lane = tid & 31, wave = tid >> 5;
  const int wm = wave & 1, wn = wave >> 1;      // 2x4 wave grid, each 64x32
  const int lane_lo = lane & 15, grp = lane >> 4;
  const int m0 = blockIdx.y * 128;              // token tile
  const int n0 = blockIdx.x * 128;              // d_out tile
  const float thr = thrp[0];

  v8f acc[4][2];
#pragma unroll
  for (int i = 0; i < 4; ++i)
#pragma unroll
    for (int j = 0; j < 2; ++j) acc[i][j] = (v8f){0.f, 0.f, 0.f, 0.f, 0.f, 0.f, 0.f, 0.f};

  const int T = DIN / TK;
  float4 pa[4], pb[4];
  g1_load(x, w, m0, n0, 0, tid, pa, pb);
  g1_store(As[0], Bs[0], thr, tid, pa, pb);
  __syncthreads();

  for (int t = 0; t < T; ++t) {
    const int cur = t & 1;
    if (t + 1 < T) g1_load(x, w, m0, n0, (t + 1) * TK, tid, pa, pb);   // prefetch
    wmma_step(As[cur], Bs[cur], wm, wn, lane_lo, grp, acc);            // compute cur
    if (t + 1 < T) {
      g1_store(As[cur ^ 1], Bs[cur ^ 1], thr, tid, pa, pb);            // fill next
      __syncthreads();
    }
  }

  const float scale = wscale[0];
#pragma unroll
  for (int mt = 0; mt < 4; ++mt) {
    const int mb = m0 + wm * 64 + mt * 16 + grp * 8;   // C/D: lanes>=16 carry M+8
#pragma unroll
    for (int nt = 0; nt < 2; ++nt) {
      const int n = n0 + wn * 32 + nt * 16 + lane_lo;
      const float bz = bias[n];
      const v8f c = acc[mt][nt];
#pragma unroll
      for (int r = 0; r < 8; ++r)
        y[(size_t)(mb + r) * DOUT + n] = c[r] * scale + bz;
    }
  }
}

// ---- GEMM2: traceO[o,i] = trace[o,i]*0.95 + (sum_b y[b,o]*x[b,i])/TOKENS ----

static __device__ __forceinline__ void g2_load(const float* __restrict__ y,
                                               const float* __restrict__ x,
                                               int m0, int n0, int kk, int tid,
                                               float4 pa[4], float4 pb[4]) {
#pragma unroll
  for (int p = 0; p < 4; ++p) {
    const int idx = p * 1024 + tid * 4;
    const int kr = idx >> 7;          // 0..31  (token row)
    const int mc = idx & 127;         // column within tile, multiple of 4
    pa[p] = *(const float4*)(y + (size_t)(kk + kr) * DOUT + m0 + mc);
    pb[p] = *(const float4*)(x + (size_t)(kk + kr) * DIN + n0 + mc);
  }
}

static __device__ __forceinline__ void g2_store(_Float16* As, _Float16* Bs, int tid,
                                                const float4 pa[4], const float4 pb[4]) {
#pragma unroll
  for (int p = 0; p < 4; ++p) {
    const int idx = p * 1024 + tid * 4;
    const int kr = idx >> 7, mc = idx & 127;
    const float4 a = pa[p];
    As[(mc + 0) * LDA + kr] = (_Float16)a.x;
    As[(mc + 1) * LDA + kr] = (_Float16)a.y;
    As[(mc + 2) * LDA + kr] = (_Float16)a.z;
    As[(mc + 3) * LDA + kr] = (_Float16)a.w;
    const float4 b = pb[p];
    Bs[(mc + 0) * LDA + kr] = (_Float16)b.x;
    Bs[(mc + 1) * LDA + kr] = (_Float16)b.y;
    Bs[(mc + 2) * LDA + kr] = (_Float16)b.z;
    Bs[(mc + 3) * LDA + kr] = (_Float16)b.w;
  }
}

__global__ __launch_bounds__(256)
void gemm2_trace(const float* __restrict__ y, const float* __restrict__ x,
                 const float* __restrict__ trace, float* __restrict__ outT) {
  __shared__ _Float16 As[2][128 * LDA];
  __shared__ _Float16 Bs[2][128 * LDA];
  const int tid = threadIdx.x;
  const int lane = tid & 31, wave = tid >> 5;
  const int wm = wave & 1, wn = wave >> 1;
  const int lane_lo = lane & 15, grp = lane >> 4;
  const int m0 = blockIdx.y * 128;              // d_out tile (rows of trace)
  const int n0 = blockIdx.x * 128;              // d_in tile (cols of trace)

  v8f acc[4][2];
#pragma unroll
  for (int i = 0; i < 4; ++i)
#pragma unroll
    for (int j = 0; j < 2; ++j) acc[i][j] = (v8f){0.f, 0.f, 0.f, 0.f, 0.f, 0.f, 0.f, 0.f};

  const int T = TOKENS / TK;
  float4 pa[4], pb[4];
  g2_load(y, x, m0, n0, 0, tid, pa, pb);
  g2_store(As[0], Bs[0], tid, pa, pb);
  __syncthreads();

  for (int t = 0; t < T; ++t) {
    const int cur = t & 1;
    if (t + 1 < T) g2_load(y, x, m0, n0, (t + 1) * TK, tid, pa, pb);   // prefetch
    wmma_step(As[cur], Bs[cur], wm, wn, lane_lo, grp, acc);            // compute cur
    if (t + 1 < T) {
      g2_store(As[cur ^ 1], Bs[cur ^ 1], tid, pa, pb);                 // fill next
      __syncthreads();
    }
  }

  const float inv_n = 1.0f / (float)TOKENS;
#pragma unroll
  for (int mt = 0; mt < 4; ++mt) {
    const int mb = m0 + wm * 64 + mt * 16 + grp * 8;
#pragma unroll
    for (int nt = 0; nt < 2; ++nt) {
      const int n = n0 + wn * 32 + nt * 16 + lane_lo;
      const v8f c = acc[mt][nt];
#pragma unroll
      for (int r = 0; r < 8; ++r) {
        const size_t off = (size_t)(mb + r) * DIN + n;
        outT[off] = trace[off] * 0.95f + c[r] * inv_n;   // TRACE_DECAY / AFFECT_GATING
      }
    }
  }
}

// ---------------------------------------------------------------------------

extern "C" void kernel_launch(void* const* d_in, const int* in_sizes, int n_in,
                              void* d_out, int out_size, void* d_ws, size_t ws_size,
                              hipStream_t stream) {
  (void)in_sizes; (void)n_in; (void)out_size; (void)ws_size;
  const float* x      = (const float*)d_in[0];   // (4096, 4096)
  const float* w      = (const float*)d_in[1];   // (16384, 4096)
  const float* bias   = (const float*)d_in[2];   // (16384,)
  const float* wscale = (const float*)d_in[3];   // (1,)
  const float* trace  = (const float*)d_in[4];   // (16384, 4096)

  float* out  = (float*)d_out;
  float* y    = out;                               // first output: (4096, 16384)
  float* outT = out + (size_t)TOKENS * DOUT;       // second output: (16384, 4096)

  float* ws       = (float*)d_ws;
  float* thr      = ws;        // [0]
  float* partials = ws + 16;   // 4096 floats

  // 1) thr = 0.1 * mean(|W|)  (deterministic two-pass tree reduction)
  absum_partial<<<4096, 256, 0, stream>>>(w, partials);
  absum_final<<<1, 256, 0, stream>>>(partials, thr);

  // 2) y = x @ ternary(W)^T * scale + bias   (WMMA f16 -> f32, double-buffered)
  gemm1_ternary<<<dim3(DOUT / 128, TOKENS / 128), 256, 0, stream>>>(x, w, bias, wscale, thr, y);

  // 3) traceO = trace*0.95 + (y^T @ x)/n     (WMMA f16 -> f32, double-buffered)
  gemm2_trace<<<dim3(DIN / 128, DOUT / 128), 256, 0, stream>>>(y, x, trace, outT);
}
